// LinearAttention2_1288490189560
// MI455X (gfx1250) — matrix operand order repaired and checked by reading, not compiled, not run on hardware
//
#include <hip/hip_runtime.h>
#include <hip/hip_bf16.h>
#include <math.h>

// ---------------------------------------------------------------------------
// Linear attention forward for MI455X (gfx1250, wave32, WMMA).
// All four GEMM stages use v_wmma_f32_16x16x32_bf16 (fp32 accumulate).
// Pipeline:
//   0) convert x -> bf16, transpose+convert w_qkv / w_proj -> bf16 [n][k]
//   1) QKV GEMM (32768x2304x768), epilogue applies elu+1 and scatters to
//      q[bh][n][d], kT[bh][d][n], vT[bh][e][n] (bf16)
//   2) per-(b,h) kv = kT @ vT^T (64x64 over K=4096), writes kv fp32 (output 2)
//      and kvT bf16 for stage 4
//   3) ksum (row sums of kT), z = 1/(q . ksum)
//   4) per-(b,h) attn = (q @ kv) * z  -> attn bf16 in (B*N, C) layout
//   5) proj GEMM (32768x768x768) + bias -> output 1 (fp32)
// ---------------------------------------------------------------------------

typedef __bf16 bf16_t;
typedef bf16_t v16bf __attribute__((ext_vector_type(16)));
typedef bf16_t v8bf  __attribute__((ext_vector_type(8)));
typedef float  v8f   __attribute__((ext_vector_type(8)));

enum { kB = 8, kN = 4096, kC = 768, kH = 12, kD = 64, kM = kB * kN, kC3 = 3 * kC };

__device__ __forceinline__ bf16_t f2bf(float f) { return (bf16_t)f; }
__device__ __forceinline__ float  bf2f(bf16_t b) { return (float)b; }
__device__ __forceinline__ float  elu1(float x) { return x > 0.f ? x + 1.f : __expf(x); }

__device__ __forceinline__ v8f zero8() {
  v8f z;
#pragma unroll
  for (int i = 0; i < 8; ++i) z[i] = 0.f;
  return z;
}

// A fragment (16x32 MxK, bf16): lane m=l%16; lanes 0-15 hold K 0-7 & 16-23,
// lanes 16-31 hold K 8-15 & 24-31 (two 16B loads, 32B apart).
__device__ __forceinline__ v16bf frag_A(const bf16_t* base, int ld) {
  const unsigned l = threadIdx.x & 31u;
  const bf16_t* p = base + (l & 15u) * ld + (l >> 4) * 8;
  union { v16bf v; struct { v8bf lo, hi; } s; } u;
  u.s.lo = *(const v8bf*)(p);
  u.s.hi = *(const v8bf*)(p + 16);
  return u.v;
}

// B fragment (32x16 KxN, bf16) loaded from BT stored [n][k]: lane n=l%16;
// lanes 0-15 hold K 0-7 & 16-23, lanes 16-31 hold K 8-15 & 24-31 (two 16B loads, 32B apart).
__device__ __forceinline__ v16bf frag_B(const bf16_t* base, int ld) {
  const unsigned l = threadIdx.x & 31u;
  const bf16_t* p = base + (l & 15u) * ld + (l >> 4) * 8;
  union { v16bf v; struct { v8bf lo, hi; } s; } u;
  u.s.lo = *(const v8bf*)(p);
  u.s.hi = *(const v8bf*)(p + 16);
  return u.v;
}

__device__ __forceinline__ v8f wmma_bf16(v16bf a, v16bf b, v8f c) {
  return __builtin_amdgcn_wmma_f32_16x16x32_bf16(false, a, false, b, (short)0, c,
                                                 false, false);
}

// ------------------------------- converters --------------------------------

__global__ __launch_bounds__(256) void k_cvt_x(const float* __restrict__ x,
                                               bf16_t* __restrict__ xb, int n) {
  int id = blockIdx.x * 256 + threadIdx.x;
  if (id < n) xb[id] = f2bf(x[id]);
}

// wT[n][k] = w[k][n] (bf16)
__global__ __launch_bounds__(256) void k_wT(const float* __restrict__ w,
                                            bf16_t* __restrict__ wT, int K, int Nw) {
  int id = blockIdx.x * 256 + threadIdx.x;
  if (id >= K * Nw) return;
  int n = id % Nw, k = id / Nw;
  wT[(size_t)n * K + k] = f2bf(w[id]);
}

// ------------------------------ QKV GEMM -----------------------------------
// C(m,j) = sum_k xb(m,k) * w(k,j);  block tile 128x128, BK=64, 8 waves.

#define TM 128
#define TN 128
#define TK 64
#define LDP 72  // padded LDS row stride (bf16 elems)

__global__ __launch_bounds__(256) void k_qkv_gemm(const bf16_t* __restrict__ xb,
                                                  const bf16_t* __restrict__ wT,
                                                  bf16_t* __restrict__ qb,
                                                  bf16_t* __restrict__ kT,
                                                  bf16_t* __restrict__ vT) {
  __shared__ __align__(16) bf16_t As[TM * LDP];
  __shared__ __align__(16) bf16_t Bs[TN * LDP];
  const int m0 = blockIdx.x * TM;
  const int n0 = blockIdx.y * TN;
  const int tid = threadIdx.x;
  const int w = tid >> 5;
  const int wm = (w & 3) * 32;   // 4 wave-rows of 32
  const int wn = (w >> 2) * 64;  // 2 wave-cols of 64

  v8f acc[2][4];
#pragma unroll
  for (int i = 0; i < 2; ++i)
#pragma unroll
    for (int j = 0; j < 4; ++j) acc[i][j] = zero8();

  for (int k0 = 0; k0 < kC; k0 += TK) {
#pragma unroll
    for (int i = 0; i < 4; ++i) {
      int chunk = tid + i * 256;          // 1024 chunks of 8 bf16 (16B)
      int row = chunk >> 3;
      int c8 = (chunk & 7) << 3;
      *(uint4*)(&As[row * LDP + c8]) =
          *(const uint4*)(&xb[(size_t)(m0 + row) * kC + k0 + c8]);
      *(uint4*)(&Bs[row * LDP + c8]) =
          *(const uint4*)(&wT[(size_t)(n0 + row) * kC + k0 + c8]);
    }
    if (k0 + TK < kC) {  // prefetch next A tile region -> global_prefetch_b8
      __builtin_prefetch(&xb[(size_t)(m0 + (tid >> 1)) * kC + k0 + TK], 0, 1);
    }
    __syncthreads();
#pragma unroll
    for (int kk = 0; kk < TK; kk += 32) {
      v16bf a0 = frag_A(&As[(wm + 0) * LDP + kk], LDP);
      v16bf a1 = frag_A(&As[(wm + 16) * LDP + kk], LDP);
#pragma unroll
      for (int nt = 0; nt < 4; ++nt) {
        v16bf bb = frag_B(&Bs[(wn + nt * 16) * LDP + kk], LDP);
        acc[0][nt] = wmma_bf16(a0, bb, acc[0][nt]);
        acc[1][nt] = wmma_bf16(a1, bb, acc[1][nt]);
      }
    }
    __syncthreads();
  }

  // Epilogue: scatter into q / kT / vT with feature map on q,k.
  const unsigned l = tid & 31u;
  const int col = l & 15, rbase = (l >> 4) * 8;
#pragma unroll
  for (int mt = 0; mt < 2; ++mt)
#pragma unroll
    for (int nt = 0; nt < 4; ++nt)
#pragma unroll
      for (int r = 0; r < 8; ++r) {
        int gm = m0 + wm + mt * 16 + rbase + r;
        int gn = n0 + wn + nt * 16 + col;
        float u = acc[mt][nt][r];
        int b = gm >> 12, ntok = gm & (kN - 1);
        int s = gn / kC;
        int rem = gn - s * kC;
        int h = rem >> 6, d = rem & 63;
        size_t bh = (size_t)(b * kH + h);
        if (s == 0)
          qb[(bh * kN + ntok) * kD + d] = f2bf(elu1(u));
        else if (s == 1)
          kT[(bh * kD + d) * kN + ntok] = f2bf(elu1(u));
        else
          vT[(bh * kD + d) * kN + ntok] = f2bf(u);
      }
}

// --------------------------- kv = kT @ vT^T --------------------------------
// One block (4 waves) per (b,h): 64x64 output, K=4096.

__global__ __launch_bounds__(128) void k_kv(const bf16_t* __restrict__ kT,
                                            const bf16_t* __restrict__ vT,
                                            float* __restrict__ kv_out,
                                            bf16_t* __restrict__ kvT) {
  const int bh = blockIdx.x;
  const int w = threadIdx.x >> 5;
  const bf16_t* A = kT + ((size_t)bh * kD + w * 16) * kN;
  const bf16_t* Bt = vT + (size_t)bh * kD * kN;
  v8f acc[4];
#pragma unroll
  for (int i = 0; i < 4; ++i) acc[i] = zero8();

  for (int ks = 0; ks < kN; ks += 32) {
    v16bf a = frag_A(A + ks, kN);
#pragma unroll
    for (int nt = 0; nt < 4; ++nt) {
      v16bf bb = frag_B(Bt + (size_t)(nt * 16) * kN + ks, kN);
      acc[nt] = wmma_bf16(a, bb, acc[nt]);
    }
  }
  const unsigned l = threadIdx.x & 31u;
  const int col = l & 15, rbase = (l >> 4) * 8;
#pragma unroll
  for (int nt = 0; nt < 4; ++nt)
#pragma unroll
    for (int r = 0; r < 8; ++r) {
      int drow = w * 16 + rbase + r;
      int e = nt * 16 + col;
      float u = acc[nt][r];
      kv_out[(size_t)bh * kD * kD + drow * kD + e] = u;       // output #2 (fp32)
      kvT[(size_t)bh * kD * kD + e * kD + drow] = f2bf(u);    // BT for stage 4
    }
}

// ------------------------ ksum and normalizer z ----------------------------

__global__ __launch_bounds__(256) void k_ksum(const bf16_t* __restrict__ kT,
                                              float* __restrict__ ksum) {
  __shared__ float red[256];
  const int row = blockIdx.x;  // bh*64 + d
  const int tid = threadIdx.x;
  float s = 0.f;
  for (int i = tid; i < kN; i += 256) s += bf2f(kT[(size_t)row * kN + i]);
  red[tid] = s;
  __syncthreads();
  for (int st = 128; st > 0; st >>= 1) {
    if (tid < st) red[tid] += red[tid + st];
    __syncthreads();
  }
  if (tid == 0) ksum[row] = red[0];
}

__global__ __launch_bounds__(256) void k_z(const bf16_t* __restrict__ qb,
                                           const float* __restrict__ ksum,
                                           float* __restrict__ z) {
  __shared__ float ks[kD];
  const int id = blockIdx.x * 256 + threadIdx.x;  // bh*4096 + ntok
  const int bh = id >> 12;
  if (threadIdx.x < kD) ks[threadIdx.x] = ksum[bh * kD + threadIdx.x];
  __syncthreads();
  const bf16_t* q = qb + (size_t)id * kD;
  float dot = 0.f;
#pragma unroll 8
  for (int d = 0; d < kD; ++d) dot += bf2f(q[d]) * ks[d];
  z[id] = 1.0f / dot;
}

// ----------------------- attn = (q @ kv) * z -------------------------------
// grid (32, 96): 128 token-rows per block, one (b,h) per blockIdx.y.

__global__ __launch_bounds__(256) void k_attn(const bf16_t* __restrict__ qb,
                                              const bf16_t* __restrict__ kvT,
                                              const float* __restrict__ z,
                                              bf16_t* __restrict__ attn) {
  const int bh = blockIdx.y;
  const int m0 = blockIdx.x * 128;
  const int w = threadIdx.x >> 5;
  const bf16_t* A = qb + ((size_t)bh * kN + m0 + w * 16) * kD;
  const bf16_t* Bt = kvT + (size_t)bh * kD * kD;
  v8f acc[4];
#pragma unroll
  for (int i = 0; i < 4; ++i) acc[i] = zero8();
#pragma unroll
  for (int kk = 0; kk < kD; kk += 32) {
    v16bf a = frag_A(A + kk, kD);
#pragma unroll
    for (int nt = 0; nt < 4; ++nt) {
      v16bf bb = frag_B(Bt + nt * 16 * kD + kk, kD);
      acc[nt] = wmma_bf16(a, bb, acc[nt]);
    }
  }
  const unsigned l = threadIdx.x & 31u;
  const int col = l & 15, rbase = (l >> 4) * 8;
  const int b = bh / kH, h = bh - b * kH;
#pragma unroll
  for (int nt = 0; nt < 4; ++nt)
#pragma unroll
    for (int r = 0; r < 8; ++r) {
      int row = m0 + w * 16 + rbase + r;
      int e = nt * 16 + col;
      float zz = z[(size_t)bh * kN + row];
      attn[((size_t)b * kN + row) * kC + h * kD + e] = f2bf(acc[nt][r] * zz);
    }
}

// ------------------------------ proj GEMM ----------------------------------

__global__ __launch_bounds__(256) void k_proj(const bf16_t* __restrict__ attn,
                                              const bf16_t* __restrict__ wT,
                                              const float* __restrict__ bias,
                                              float* __restrict__ out) {
  __shared__ __align__(16) bf16_t As[TM * LDP];
  __shared__ __align__(16) bf16_t Bs[TN * LDP];
  const int m0 = blockIdx.x * TM;
  const int n0 = blockIdx.y * TN;
  const int tid = threadIdx.x;
  const int w = tid >> 5;
  const int wm = (w & 3) * 32;
  const int wn = (w >> 2) * 64;

  v8f acc[2][4];
#pragma unroll
  for (int i = 0; i < 2; ++i)
#pragma unroll
    for (int j = 0; j < 4; ++j) acc[i][j] = zero8();

  for (int k0 = 0; k0 < kC; k0 += TK) {
#pragma unroll
    for (int i = 0; i < 4; ++i) {
      int chunk = tid + i * 256;
      int row = chunk >> 3;
      int c8 = (chunk & 7) << 3;
      *(uint4*)(&As[row * LDP + c8]) =
          *(const uint4*)(&attn[(size_t)(m0 + row) * kC + k0 + c8]);
      *(uint4*)(&Bs[row * LDP + c8]) =
          *(const uint4*)(&wT[(size_t)(n0 + row) * kC + k0 + c8]);
    }
    if (k0 + TK < kC) {
      __builtin_prefetch(&attn[(size_t)(m0 + (tid >> 1)) * kC + k0 + TK], 0, 1);
    }
    __syncthreads();
#pragma unroll
    for (int kk = 0; kk < TK; kk += 32) {
      v16bf a0 = frag_A(&As[(wm + 0) * LDP + kk], LDP);
      v16bf a1 = frag_A(&As[(wm + 16) * LDP + kk], LDP);
#pragma unroll
      for (int nt = 0; nt < 4; ++nt) {
        v16bf bb = frag_B(&Bs[(wn + nt * 16) * LDP + kk], LDP);
        acc[0][nt] = wmma_bf16(a0, bb, acc[0][nt]);
        acc[1][nt] = wmma_bf16(a1, bb, acc[1][nt]);
      }
    }
    __syncthreads();
  }
  const unsigned l = tid & 31u;
  const int col = l & 15, rbase = (l >> 4) * 8;
#pragma unroll
  for (int mt = 0; mt < 2; ++mt)
#pragma unroll
    for (int nt = 0; nt < 4; ++nt)
#pragma unroll
      for (int r = 0; r < 8; ++r) {
        int gm = m0 + wm + mt * 16 + rbase + r;
        int gn = n0 + wn + nt * 16 + col;
        out[(size_t)gm * kC + gn] = acc[mt][nt][r] + bias[gn];
      }
}

// ------------------------------ launcher -----------------------------------

extern "C" void kernel_launch(void* const* d_in, const int* in_sizes, int n_in,
                              void* d_out, int out_size, void* d_ws, size_t ws_size,
                              hipStream_t stream) {
  const float* x      = (const float*)d_in[0];
  const float* w_qkv  = (const float*)d_in[1];
  const float* w_proj = (const float*)d_in[2];
  const float* b_proj = (const float*)d_in[3];

  size_t off = 0;
  auto alloc = [&](size_t bytes) -> void* {
    void* p = (char*)d_ws + off;
    off += (bytes + 255) & ~(size_t)255;
    return p;
  };
  bf16_t* xb     = (bf16_t*)alloc((size_t)kM * kC * 2);          // x bf16
  bf16_t* wqkvT  = (bf16_t*)alloc((size_t)kC3 * kC * 2);         // [n][k]
  bf16_t* wprojT = (bf16_t*)alloc((size_t)kC * kC * 2);          // [n][k]
  bf16_t* qb     = (bf16_t*)alloc((size_t)kB * kH * kN * kD * 2); // [bh][n][d]
  bf16_t* kT     = (bf16_t*)alloc((size_t)kB * kH * kD * kN * 2); // [bh][d][n]
  bf16_t* vT     = (bf16_t*)alloc((size_t)kB * kH * kD * kN * 2); // [bh][e][n]
  bf16_t* kvT    = (bf16_t*)alloc((size_t)kB * kH * kD * kD * 2); // [bh][e][d]
  float*  ksum   = (float*) alloc((size_t)kB * kH * kD * 4);
  float*  zbuf   = (float*) alloc((size_t)kB * kH * kN * 4);
  bf16_t* attn   = (bf16_t*)alloc((size_t)kM * kC * 2);          // [m][c]

  float* out_main = (float*)d_out;                 // (B,N,C)
  float* out_kv   = (float*)d_out + (size_t)kM * kC;  // (B,H,D,D)

  // 0) converts / transposes
  k_cvt_x<<<(kM * kC + 255) / 256, 256, 0, stream>>>(x, xb, kM * kC);
  k_wT<<<(kC * kC3 + 255) / 256, 256, 0, stream>>>(w_qkv, wqkvT, kC, kC3);
  k_wT<<<(kC * kC + 255) / 256, 256, 0, stream>>>(w_proj, wprojT, kC, kC);

  // 1) QKV projection + feature map + layout scatter
  k_qkv_gemm<<<dim3(kM / TM, kC3 / TN), 256, 0, stream>>>(xb, wqkvT, qb, kT, vT);

  // 2) kv per head (also writes fp32 kv output)
  k_kv<<<kB * kH, 128, 0, stream>>>(kT, vT, out_kv, kvT);

  // 3) ksum and z
  k_ksum<<<kB * kH * kD, 256, 0, stream>>>(kT, ksum);
  k_z<<<(kB * kH * kN) / 256, 256, 0, stream>>>(qb, ksum, zbuf);

  // 4) attn = (q @ kv) * z
  k_attn<<<dim3(kN / 128, kB * kH), 256, 0, stream>>>(qb, kvT, zbuf, attn);

  // 5) output projection + bias
  k_proj<<<dim3(kM / TM, kC / TN), 256, 0, stream>>>(attn, wprojT, b_proj, out_main);

  (void)in_sizes; (void)n_in; (void)out_size; (void)ws_size;
}
